// GRU_30537217475004
// MI455X (gfx1250) — matrix operand order repaired and checked by reading, not compiled, not run on hardware
//
#include <hip/hip_runtime.h>
#include <hip/hip_bf16.h>

// GRU on MI455X (gfx1250), wave32 + WMMA bf16.
// Phase 0: convert W_ih / W_hh to bf16 in ws.
// Phase 1: gi = x @ W_ih^T (+ b_ih, + b_hh for r,z gates), bf16 WMMA,
//          stored to ws pre-swizzled in WMMA C-fragment layout.
// Phase 2: sequential scan. 4 workgroups (batch-split, no inter-WG sync),
//          16 waves each; W_hh resident in VGPRs, h double-buffered in LDS,
//          gi prefetched one timestep ahead.

typedef __bf16 bf16_t;
typedef __attribute__((ext_vector_type(16))) bf16_t        v16bf;
typedef __attribute__((ext_vector_type(8)))  float         v8f;
typedef __attribute__((ext_vector_type(4)))  unsigned int  u32x4;
typedef __attribute__((ext_vector_type(8)))  unsigned short u16x8;
typedef __attribute__((ext_vector_type(4)))  float         f32x4;

#define T_STEPS 2048
#define BATCH   64
#define HID     256
#define GATES   768                       // 3*H
#define MTILES  (T_STEPS * BATCH / 16)    // 8192 row tiles of 16
#define HROW    264                       // LDS h row stride (256 + 8 pad), bf16 units

__device__ __forceinline__ unsigned short f2bf(float f) {
  unsigned int u = __builtin_bit_cast(unsigned int, f);
  u += 0x7FFFu + ((u >> 16) & 1u);        // round-to-nearest-even
  return (unsigned short)(u >> 16);
}
__device__ __forceinline__ float bf2f(unsigned short s) {
  unsigned int u = ((unsigned int)s) << 16;
  return __builtin_bit_cast(float, u);
}
__device__ __forceinline__ float fsig(float x) {
  return __builtin_amdgcn_rcpf(1.0f + __builtin_amdgcn_exp2f(-1.442695041f * x));
}
__device__ __forceinline__ float ftanh(float x) {
  return 2.0f * fsig(2.0f * x) - 1.0f;
}

union Frag16 { v16bf bf; u32x4 q[2]; unsigned short s[16]; };
union Acc    { v8f f; float e[8]; };
union Pack8  { u16x8 v; unsigned short s[8]; };

// ---------------- Phase 0: f32 -> bf16 weight conversion ----------------
__global__ void __launch_bounds__(256)
cvt_weights(const float* __restrict__ wih, const float* __restrict__ whh,
            unsigned short* __restrict__ dst) {
  int i = blockIdx.x * 256 + threadIdx.x;            // 0 .. 2*GATES*HID-1
  float v = (i < GATES * HID) ? wih[i] : whh[i - GATES * HID];
  dst[i] = f2bf(v);
}

// ---------------- Phase 1: gi = x @ W_ih^T + biases (swizzled bf16) -----
// One workgroup (8 waves) per 16-row tile mt: t = mt/4, b0 = (mt%4)*16.
// Wave wv computes column tiles c = 6*wv .. 6*wv+5 (48 tiles of N=16).
__global__ void __launch_bounds__(256)
gru_phase1(const float* __restrict__ x,
           const float* __restrict__ bias_ih,
           const float* __restrict__ bias_hh,
           const unsigned short* __restrict__ wih_bf,
           unsigned short* __restrict__ gi) {
  const int mt   = blockIdx.x;
  const int t    = mt >> 2;
  const int b0   = (mt & 3) << 4;
  const int lane = threadIdx.x & 31;
  const int wv   = threadIdx.x >> 5;     // 0..7
  const int m    = lane & 15;
  const int kh   = lane >> 4;            // K-half select per ISA A layout

  // A fragments: x row (b0+m, t, :) converted to bf16, ISA 16-bit 16x32 layout:
  // lane holds K runs [kb, kb+8) and [kb+16, kb+24), kb = kk*32 + kh*8.
  Frag16 A[8];
  const float* xrow = x + ((size_t)(b0 + m) * T_STEPS + t) * 256;
#pragma unroll
  for (int kk = 0; kk < 8; ++kk) {
    const int kb = kk * 32 + kh * 8;
    f32x4 a0 = *(const f32x4*)(xrow + kb);
    f32x4 a1 = *(const f32x4*)(xrow + kb + 4);
    f32x4 a2 = *(const f32x4*)(xrow + kb + 16);
    f32x4 a3 = *(const f32x4*)(xrow + kb + 20);
#pragma unroll
    for (int e = 0; e < 4; ++e) {
      A[kk].s[e]      = f2bf(a0[e]);
      A[kk].s[4 + e]  = f2bf(a1[e]);
      A[kk].s[8 + e]  = f2bf(a2[e]);
      A[kk].s[12 + e] = f2bf(a3[e]);
    }
  }

  for (int ci = 0; ci < 6; ++ci) {
    const int c = wv * 6 + ci;                       // column tile 0..47
    Acc acc;
#pragma unroll
    for (int v = 0; v < 8; ++v) acc.e[v] = 0.0f;
#pragma unroll
    for (int kk = 0; kk < 8; ++kk) {
      // B 32x16 bf16 layout: lane = (kh,n); per lane K runs [kh*8, kh*8+8) and
      // [16+kh*8, 24+kh*8) of W_ih row n = c*16+m  (W row-major == B column-major).
      Frag16 B;
      const unsigned short* bp =
          wih_bf + (size_t)(c * 16 + m) * 256 + kk * 32 + kh * 8;
      B.q[0] = *(const u32x4*)bp;
      B.q[1] = *(const u32x4*)(bp + 16);
      acc.f = __builtin_amdgcn_wmma_f32_16x16x32_bf16(
          false, A[kk].bf, false, B.bf, (short)0, acc.f, false, false);
    }
    // Fold biases: b_ih always; b_hh for r,z gates (tiles 0..31) only.
    const int col = c * 16 + m;
    const float bias = bias_ih[col] + ((c < 32) ? bias_hh[col] : 0.0f);
    Pack8 p;
#pragma unroll
    for (int v = 0; v < 8; ++v) p.s[v] = f2bf(acc.e[v] + bias);
    // Pre-swizzled store: C-fragment layout, 16B contiguous per lane.
    unsigned short* dst = gi + ((size_t)mt * 48 + c) * 256 + lane * 8;
    *(u16x8*)dst = p.v;
  }
}

// ---------------- Phase 2: sequential GRU scan --------------------------
// Block wg handles batch rows [16*wg, 16*wg+16). 16 waves; wave wv owns
// hidden cols [16*wv, 16*wv+16) for all three gates (tiles wv, 16+wv, 32+wv).
__global__ void __launch_bounds__(512)
gru_phase2(const unsigned short* __restrict__ whh_bf,
           const unsigned short* __restrict__ gi,
           const float* __restrict__ bias_hh,
           float* __restrict__ out) {
  __shared__ __align__(16) unsigned short hbuf[2][16 * HROW];
  const int lane = threadIdx.x & 31;
  const int wv   = threadIdx.x >> 5;    // 0..15
  const int wg   = blockIdx.x;          // 0..3
  const int m    = lane & 15;
  const int kh   = lane >> 4;

  // Resident W_hh fragments: 3 gates x 8 k-frags = 192 VGPRs/wave.
  Frag16 W[3][8];
#pragma unroll
  for (int g = 0; g < 3; ++g)
#pragma unroll
    for (int kk = 0; kk < 8; ++kk) {
      const unsigned short* bp =
          whh_bf + (size_t)((g * 16 + wv) * 16 + m) * 256 + kk * 32 + kh * 8;
      W[g][kk].q[0] = *(const u32x4*)bp;
      W[g][kk].q[1] = *(const u32x4*)(bp + 16);
    }
  const float bn = bias_hh[2 * HID + wv * 16 + m];  // n-gate recurrent bias

  for (int i = threadIdx.x; i < 16 * HROW; i += 512) hbuf[0][i] = 0;
  __syncthreads();

  // Software pipeline: gi tiles for step t prefetched during step t-1.
  u16x8 giA[3], giB[3];
#pragma unroll
  for (int g = 0; g < 3; ++g)
    giA[g] = *(const u16x8*)(gi + ((size_t)wg * 48 + (g * 16 + wv)) * 256 +
                             (size_t)lane * 8);

  auto step = [&](int t, u16x8(&cg)[3], u16x8(&ng)[3]) {
    const int tn = (t + 1 < T_STEPS) ? (t + 1) : t;
#pragma unroll
    for (int g = 0; g < 3; ++g)
      ng[g] = *(const u16x8*)(gi +
                ((size_t)(tn * 4 + wg) * 48 + (g * 16 + wv)) * 256 +
                (size_t)lane * 8);

    const unsigned short* hc = &hbuf[t & 1][0];
    unsigned short*       hw = &hbuf[(t & 1) ^ 1][0];

    float hprev[8];
#pragma unroll
    for (int v = 0; v < 8; ++v)
      hprev[v] = bf2f(hc[(v + 8 * kh) * HROW + wv * 16 + m]);

    Acc aR, aZ, aN;
#pragma unroll
    for (int v = 0; v < 8; ++v) { aR.e[v] = 0.0f; aZ.e[v] = 0.0f; aN.e[v] = 0.0f; }
#pragma unroll
    for (int kk = 0; kk < 8; ++kk) {
      Frag16 Af;                                    // A from LDS h, 2x b128
      const unsigned short* ap = hc + m * HROW + kk * 32 + kh * 8;
      Af.q[0] = *(const u32x4*)ap;
      Af.q[1] = *(const u32x4*)(ap + 16);
      aR.f = __builtin_amdgcn_wmma_f32_16x16x32_bf16(
          false, Af.bf, false, W[0][kk].bf, (short)0, aR.f, false, false);
      aZ.f = __builtin_amdgcn_wmma_f32_16x16x32_bf16(
          false, Af.bf, false, W[1][kk].bf, (short)0, aZ.f, false, false);
      aN.f = __builtin_amdgcn_wmma_f32_16x16x32_bf16(
          false, Af.bf, false, W[2][kk].bf, (short)0, aN.f, false, false);
    }

#pragma unroll
    for (int v = 0; v < 8; ++v) {
      float r  = fsig(bf2f(cg[0][v]) + aR.e[v]);          // i_r(+biases)+h_r
      float z  = fsig(bf2f(cg[1][v]) + aZ.e[v]);          // i_z(+biases)+h_z
      float n  = ftanh(bf2f(cg[2][v]) + r * (aN.e[v] + bn));
      float hv = n + z * (hprev[v] - n);                  // (1-z)*n + z*h
      hw[(v + 8 * kh) * HROW + wv * 16 + m] = f2bf(hv);
      if (t == T_STEPS - 1)
        out[(size_t)(wg * 16 + v + 8 * kh) * HID + wv * 16 + m] = hv;
    }
    __syncthreads();   // double-buffered h: one barrier per step
  };

  for (int t = 0; t < T_STEPS; t += 2) {
    step(t, giA, giB);
    step(t + 1, giB, giA);
  }
}

// ---------------- host launch -------------------------------------------
extern "C" void kernel_launch(void* const* d_in, const int* in_sizes, int n_in,
                              void* d_out, int out_size, void* d_ws, size_t ws_size,
                              hipStream_t stream) {
  const float* x   = (const float*)d_in[0];   // [64,2048,256]
  const float* wih = (const float*)d_in[1];   // [768,256]
  const float* whh = (const float*)d_in[2];   // [768,256]
  const float* bih = (const float*)d_in[3];   // [768]
  const float* bhh = (const float*)d_in[4];   // [768]

  // ws layout (bf16 as u16): [W_ih 196608][W_hh 196608][gi 8192*48*256]
  // total ~202.1 MB.
  unsigned short* wih_bf = (unsigned short*)d_ws;
  unsigned short* whh_bf = wih_bf + (size_t)GATES * HID;
  unsigned short* gi     = whh_bf + (size_t)GATES * HID;

  cvt_weights<<<dim3((2 * GATES * HID) / 256), dim3(256), 0, stream>>>(
      wih, whh, wih_bf);
  gru_phase1<<<dim3(MTILES), dim3(256), 0, stream>>>(
      x, bih, bhh, wih_bf, gi);
  gru_phase2<<<dim3(BATCH / 16), dim3(512), 0, stream>>>(
      whh_bf, gi, bhh, (float*)d_out);
}
